// Criterion_24773371363769
// MI455X (gfx1250) — matrix operand order repaired and checked by reading, not compiled, not run on hardware
//
#include <hip/hip_runtime.h>
#include <math.h>

// ---------------- problem constants ----------------
#define BB   4096   // batch rows
#define DD   1024   // embedding dim
#define NA   2048   // anchors
#define NNEG 4094   // negatives per anchor
#define L2W  0.005f

// ---------------- CDNA5 WMMA / TDM types ----------------
typedef __bf16 v16bf __attribute__((ext_vector_type(16)));
typedef __bf16 v8bf  __attribute__((ext_vector_type(8)));
typedef float  v8f   __attribute__((ext_vector_type(8)));
typedef unsigned int u32x4 __attribute__((ext_vector_type(4)));
typedef int          i32x8 __attribute__((ext_vector_type(8)));
typedef int          i32x4 __attribute__((ext_vector_type(4)));

__device__ __forceinline__ v8f wmma_bf16(v16bf a, v16bf b, v8f c) {
  // D = A(16x32) * B(32x16) + C(16x16), f32 accumulate
  return __builtin_amdgcn_wmma_f32_16x16x32_bf16(
      /*neg_a=*/false, a, /*neg_b=*/false, b,
      /*c_mod=*/(short)0, c, /*reuse_a=*/false, /*reuse_b=*/false);
}

// Issue a 2D TDM load: tile 128(N, contiguous) x 32(K rows) of bf16 from a
// [1024][4096] row-major tensor into LDS (linear [32][128] layout).
// D# fields per CDNA5 ISA 8.3/8.4: group0 = {count/type/lds/global addr},
// group1 = {data_size, tensor dims, tile dims, strides}. 6-arg builtin form
// (clang-23 / therock-10.0): (g0, g1, g2, g3, g_extra, cpol).
__device__ __forceinline__ void tdm_load_b_tile(unsigned lds_off, const void* gsrc) {
  unsigned long long ga = (unsigned long long)(uintptr_t)gsrc;
  u32x4 g0;
  g0[0] = 1u;                                     // count=1, user descriptor
  g0[1] = lds_off;                                // lds_addr (bytes)
  g0[2] = (unsigned)(ga & 0xFFFFFFFFu);           // global_addr[31:0]
  g0[3] = (unsigned)((ga >> 32) & 0x01FFFFFFu)    // global_addr[56:32]
        | (2u << 30);                             // type = 2 ("image")
  i32x8 g1;
  g1[0] = (int)(1u << 16);        // workgroup_mask=0, data_size=1 (2 bytes)
  g1[1] = (int)(4096u << 16);     // tensor_dim0[15:0]=4096 (bits 79:48)
  g1[2] = (int)(1024u << 16);     // tensor_dim0 hi=0 | tensor_dim1[15:0]=1024
  g1[3] = (int)(128u << 16);      // tensor_dim1 hi=0 | tile_dim0=128
  g1[4] = 32;                     // tile_dim1=32, tile_dim2=0
  g1[5] = 4096;                   // tensor_dim0_stride lo (elements)
  g1[6] = 0;                      // stride hi / tensor_dim1_stride lo
  g1[7] = 0;
  i32x4 z4 = {0, 0, 0, 0};        // groups 2/3 unused (2D tensor)
  i32x8 z8 = {0, 0, 0, 0, 0, 0, 0, 0};
  __builtin_amdgcn_tensor_load_to_lds(g0, g1, z4, z4, z8, 0);
}

// ---------------- K0: zero the two scalar accumulators ----------------
__global__ void k_init(float* __restrict__ accum) {
  accum[0] = 0.0f;  // sum of row norms
  accum[1] = 0.0f;  // sum of per-anchor log terms / NA
}

// ---------------- K1: mean row-norm term ----------------
__global__ __launch_bounds__(256)
void k_row_norms(const float* __restrict__ batch, float* __restrict__ accum) {
  __shared__ float red[256];
  const int row = blockIdx.x;
  float s = 0.0f;
  for (int c = threadIdx.x; c < DD; c += 256) {
    float x = batch[(size_t)row * DD + c];
    s += x * x;
  }
  red[threadIdx.x] = s;
  __syncthreads();
  for (int off = 128; off > 0; off >>= 1) {
    if ((int)threadIdx.x < off) red[threadIdx.x] += red[threadIdx.x + off];
    __syncthreads();
  }
  if (threadIdx.x == 0) atomicAdd(&accum[0], sqrtf(red[0]));
}

// ---------------- K2: row-major split-bf16 convert (A-side source) --------
__global__ __launch_bounds__(256)
void k_convert_rm(const float* __restrict__ batch,
                  __bf16* __restrict__ bhi, __bf16* __restrict__ blo) {
  size_t i = (size_t)blockIdx.x * 256 + threadIdx.x;  // exactly B*D threads
  float x = batch[i];
  __bf16 h = (__bf16)x;
  bhi[i] = h;
  blo[i] = (__bf16)(x - (float)h);
}

// ---------------- K3: tiled transpose + split-bf16 convert (B-side) -------
__global__ __launch_bounds__(256)
void k_transpose_convert(const float* __restrict__ batch,
                         __bf16* __restrict__ bhiT, __bf16* __restrict__ bloT) {
  __shared__ __bf16 shi[64][66];
  __shared__ __bf16 slo[64][66];
  const int c0 = blockIdx.x * 64;          // along D
  const int r0 = blockIdx.y * 64;          // along B
  const int tx = threadIdx.x & 63;
  const int ty = threadIdx.x >> 6;         // 0..3
#pragma unroll
  for (int i = 0; i < 16; ++i) {
    int r = ty + i * 4;
    float x = batch[(size_t)(r0 + r) * DD + c0 + tx];
    __bf16 h = (__bf16)x;
    shi[r][tx] = h;
    slo[r][tx] = (__bf16)(x - (float)h);
  }
  __syncthreads();
#pragma unroll
  for (int i = 0; i < 16; ++i) {
    int c = ty + i * 4;
    bhiT[(size_t)(c0 + c) * BB + r0 + tx] = shi[tx][c];
    bloT[(size_t)(c0 + c) * BB + r0 + tx] = slo[tx][c];
  }
}

// ---------------- K4: split-bf16 WMMA GEMM with LDS + TDM staging ---------
// Workgroup = 4 waves computing a 128(M)x128(N) block; wave (mq,nq) owns a
// 64x64 quadrant. Per K=32 step: A chunk (128x32, anchor-gathered) staged by
// all 128 threads via b128 loads + ds stores; B chunk (32x128, uniform) DMA'd
// by the Tensor Data Mover. Double-buffered, fenced with s_wait_tensorcnt +
// workgroup barrier.
__global__ __launch_bounds__(128)
void k_gemm_scores(const __bf16* __restrict__ bhi,  const __bf16* __restrict__ blo,
                   const __bf16* __restrict__ bhiT, const __bf16* __restrict__ bloT,
                   const int* __restrict__ anchors, float* __restrict__ scores) {
  __shared__ __attribute__((aligned(64))) __bf16 sAhi[2][128][32];
  __shared__ __attribute__((aligned(64))) __bf16 sAlo[2][128][32];
  __shared__ __attribute__((aligned(64))) __bf16 sBhi[2][32][128];
  __shared__ __attribute__((aligned(64))) __bf16 sBlo[2][32][128];

  const int tid  = threadIdx.x;
  const int lane = tid & 31;
  const int wav  = tid >> 5;       // 0..3
  const int mq   = wav >> 1;       // M quadrant (0..1)
  const int nq   = wav & 1;        // N quadrant (0..1)
  const int half = lane >> 4;
  const int l16  = lane & 15;
  const int m0 = blockIdx.y * 128;
  const int n0 = blockIdx.x * 128;

  // each thread stages one anchor-gathered A row per chunk
  const int myrow = anchors[m0 + tid];
  const __bf16* arow_hi = bhi + (size_t)myrow * DD;
  const __bf16* arow_lo = blo + (size_t)myrow * DD;

  v8f acc[4][4];
  const v8f vzero = {0.f, 0.f, 0.f, 0.f, 0.f, 0.f, 0.f, 0.f};
#pragma unroll
  for (int mt = 0; mt < 4; ++mt)
#pragma unroll
    for (int nt = 0; nt < 4; ++nt) acc[mt][nt] = vzero;

  // ---- prologue: stage k-chunk 0 into buffer 0 ----
  {
    *(v16bf*)&sAhi[0][tid][0]  = *(const v16bf*)(arow_hi + 0);
    *(v16bf*)&sAhi[0][tid][16] = *(const v16bf*)(arow_hi + 16);
    *(v16bf*)&sAlo[0][tid][0]  = *(const v16bf*)(arow_lo + 0);
    *(v16bf*)&sAlo[0][tid][16] = *(const v16bf*)(arow_lo + 16);
    if (wav == 0) {
      tdm_load_b_tile((unsigned)(uintptr_t)&sBhi[0][0][0], bhiT + (size_t)0 * BB + n0);
      tdm_load_b_tile((unsigned)(uintptr_t)&sBlo[0][0][0], bloT + (size_t)0 * BB + n0);
      __builtin_amdgcn_s_wait_tensorcnt(0);
    }
  }
  __syncthreads();

  int buf = 0;
  for (int kstep = 0; kstep < DD / 32; ++kstep) {
    const int nbuf = buf ^ 1;
    // ---- stage next k-chunk into the other buffer while computing ----
    if (kstep + 1 < DD / 32) {
      const int kn = (kstep + 1) * 32;
      *(v16bf*)&sAhi[nbuf][tid][0]  = *(const v16bf*)(arow_hi + kn);
      *(v16bf*)&sAhi[nbuf][tid][16] = *(const v16bf*)(arow_hi + kn + 16);
      *(v16bf*)&sAlo[nbuf][tid][0]  = *(const v16bf*)(arow_lo + kn);
      *(v16bf*)&sAlo[nbuf][tid][16] = *(const v16bf*)(arow_lo + kn + 16);
      if (wav == 0) {
        tdm_load_b_tile((unsigned)(uintptr_t)&sBhi[nbuf][0][0],
                        bhiT + (size_t)kn * BB + n0);
        tdm_load_b_tile((unsigned)(uintptr_t)&sBlo[nbuf][0][0],
                        bloT + (size_t)kn * BB + n0);
      }
    }

    // ---- compute current buffer ----
    // A operand (16x32 bf16): lane l16 = M row; elems 0..7 = K ka..ka+7,
    // elems 8..15 = K ka+16..ka+23, ka = half*8.
    v16bf ahi[4], alo[4];
#pragma unroll
    for (int mt = 0; mt < 4; ++mt) {
      const __bf16* ph = &sAhi[buf][mq * 64 + mt * 16 + l16][half * 8];
      const __bf16* pl = &sAlo[buf][mq * 64 + mt * 16 + l16][half * 8];
      v8bf h0 = *(const v8bf*)(ph);
      v8bf h1 = *(const v8bf*)(ph + 16);
      v8bf l0 = *(const v8bf*)(pl);
      v8bf l1 = *(const v8bf*)(pl + 16);
      ahi[mt] = __builtin_shufflevector(h0, h1, 0,1,2,3,4,5,6,7,8,9,10,11,12,13,14,15);
      alo[mt] = __builtin_shufflevector(l0, l1, 0,1,2,3,4,5,6,7,8,9,10,11,12,13,14,15);
    }
    // B operand (32x16 bf16): lane l16 = N col; elems 0..7 = K ka..ka+7,
    // elems 8..15 = K ka+16..ka+23, ka = half*8.
    v16bf bh[4], bl[4];
#pragma unroll
    for (int nt = 0; nt < 4; ++nt) {
#pragma unroll
      for (int s = 0; s < 16; ++s) {
        const int kk = (s < 8) ? (half * 8 + s) : (16 + half * 8 + (s - 8));
        bh[nt][s] = sBhi[buf][kk][nq * 64 + nt * 16 + l16];
        bl[nt][s] = sBlo[buf][kk][nq * 64 + nt * 16 + l16];
      }
    }
#pragma unroll
    for (int mt = 0; mt < 4; ++mt)
#pragma unroll
      for (int nt = 0; nt < 4; ++nt) {
        acc[mt][nt] = wmma_bf16(ahi[mt], bh[nt], acc[mt][nt]);
        acc[mt][nt] = wmma_bf16(ahi[mt], bl[nt], acc[mt][nt]);
        acc[mt][nt] = wmma_bf16(alo[mt], bh[nt], acc[mt][nt]);
      }

    // ---- fence staging of nbuf before it becomes current ----
    if (wav == 0) __builtin_amdgcn_s_wait_tensorcnt(0);
    __syncthreads();
    buf = nbuf;
  }

  // C/D layout: lanes 0-15 => N = lane, VGPR r => M = r; lanes 16-31 => M = r+8
#pragma unroll
  for (int mt = 0; mt < 4; ++mt)
#pragma unroll
    for (int nt = 0; nt < 4; ++nt)
#pragma unroll
      for (int r = 0; r < 8; ++r) {
        int m = m0 + mq * 64 + mt * 16 + half * 8 + r;
        int n = n0 + nq * 64 + nt * 16 + l16;
        scores[(size_t)m * BB + n] = acc[mt][nt][r];
      }
}

// ---------------- K5: per-anchor gather + log(sum(exp)+1) ----------------
__global__ __launch_bounds__(256)
void k_npair(const float* __restrict__ scores, const int* __restrict__ positives,
             const int* __restrict__ negatives, float* __restrict__ accum) {
  __shared__ float red[256];
  const int i = blockIdx.x;
  const float* row = scores + (size_t)i * BB;
  const float pos = row[positives[i]];
  const int* neg = negatives + (size_t)i * NNEG;
  float s = 0.0f;
  for (int t = threadIdx.x; t < NNEG; t += 256)
    s += expf(row[neg[t]] - pos);
  red[threadIdx.x] = s;
  __syncthreads();
  for (int off = 128; off > 0; off >>= 1) {
    if ((int)threadIdx.x < off) red[threadIdx.x] += red[threadIdx.x + off];
    __syncthreads();
  }
  if (threadIdx.x == 0)
    atomicAdd(&accum[1], logf(red[0] + 1.0f) * (1.0f / (float)NA));
}

// ---------------- K6: finalize scalar ----------------
__global__ void k_finalize(const float* __restrict__ accum, float* __restrict__ out) {
  out[0] = accum[1] + L2W * (accum[0] * (1.0f / (float)BB));
}

// ---------------- launch ----------------
extern "C" void kernel_launch(void* const* d_in, const int* in_sizes, int n_in,
                              void* d_out, int out_size, void* d_ws, size_t ws_size,
                              hipStream_t stream) {
  const float* batch     = (const float*)d_in[0];
  const int*   anchors   = (const int*)d_in[1];
  const int*   positives = (const int*)d_in[2];
  const int*   negatives = (const int*)d_in[3];
  float* out = (float*)d_out;

  char* w = (char*)d_ws;
  const size_t MB = 1024ull * 1024ull;
  __bf16* bhi    = (__bf16*)(w + 0 * MB);    // [B][D]  8 MB
  __bf16* blo    = (__bf16*)(w + 8 * MB);    // [B][D]  8 MB
  __bf16* bhiT   = (__bf16*)(w + 16 * MB);   // [D][B]  8 MB
  __bf16* bloT   = (__bf16*)(w + 24 * MB);   // [D][B]  8 MB
  float*  scores = (float*)(w + 32 * MB);    // [NA][B] 32 MB (L2-resident)
  float*  accum  = (float*)(w + 64 * MB);    // 2 floats

  k_init<<<1, 1, 0, stream>>>(accum);
  k_row_norms<<<BB, 256, 0, stream>>>(batch, accum);
  k_convert_rm<<<(BB * DD) / 256, 256, 0, stream>>>(batch, bhi, blo);
  k_transpose_convert<<<dim3(DD / 64, BB / 64), 256, 0, stream>>>(batch, bhiT, bloT);
  k_gemm_scores<<<dim3(BB / 128, NA / 128), 128, 0, stream>>>(bhi, blo, bhiT, bloT,
                                                              anchors, scores);
  k_npair<<<NA, 256, 0, stream>>>(scores, positives, negatives, accum);
  k_finalize<<<1, 1, 0, stream>>>(accum, out);
}
